// Encoder_2697239462473
// MI455X (gfx1250) — matrix operand order repaired and checked by reading, not compiled, not run on hardware
//
#include <hip/hip_runtime.h>
#include <hip/hip_bf16.h>
#include <stdint.h>

// Bidirectional GRU encoder for MI455X (gfx1250, wave32, WMMA).
//  1) pack_b    : f32 weights -> bf16 WMMA-B tiles (padded-N' layout) for proj
//  2) pack_scan : f32 recurrent weights -> bf16 [group][kt][gate] tiles for scan
//  3) proj      : xp = x @ K + b_in; 3 ntile chunks x double-buffered kt loads
//                 feed 3 independent WMMA chains; NT stores (stream-once xp)
//  4) scan      : 8 persistent workgroups run the 1024-step recurrence; h in f32
//                 registers + bf16 LDS shadow (WMMA A); double-buffered B loads;
//                 global_prefetch of xp two steps ahead; 2 barriers per step
//  5) fc        : state = relu([h_f|h_b] @ w_fc + b_fc)

#define B_  64
#define T_  1024
#define D_  300
#define U_  300
#define N3  900
#define NP  912   // padded N': [0,300) z | gap | [304,604) r | gap | [608,908) h | gap
#define NT  57    // NP/16 n-tiles (= 19 chunks of 3)
#define KT  10    // K padded 300 -> 320
#define KP  320
#define NG  19    // unit groups of 16 (19*16 = 304 >= 300)

typedef __attribute__((ext_vector_type(16))) __bf16 v16bf;
typedef __attribute__((ext_vector_type(8)))  __bf16 v8bf;
typedef __attribute__((ext_vector_type(8)))  float  v8f;

static __device__ __forceinline__ int np_to_n(int np) {
  if (np < 300) return np;
  if (np < 304) return -1;
  if (np < 604) return np - 4;
  if (np < 608) return -1;
  if (np < 908) return np - 8;
  return -1;
}

// ---------------------------------------------------------------------------
// Pack (300 x 900) f32 row-major -> bf16 WMMA-B tiles for proj:
// dst[nt][kt][lane][16]; lanes 0-15 hold K=kt*32+{0-7,16-23}, lanes 16-31 K=kt*32+{8-15,24-31},
// column = nt*16 + (lane&15) in padded-N' space.
// ---------------------------------------------------------------------------
__global__ void pack_b(const float* __restrict__ src, __bf16* __restrict__ dst) {
  int idx = blockIdx.x * 256 + threadIdx.x;   // one thread per (nt,kt,lane)
  if (idx >= NT * KT * 32) return;
  int lane = idx & 31;
  int kt   = (idx >> 5) % KT;
  int nt   = idx / (32 * KT);
  int npc  = nt * 16 + (lane & 15);
  int n    = np_to_n(npc);
  int kb   = kt * 32 + ((lane < 16) ? 0 : 8);
  __bf16* o = dst + (size_t)idx * 16;
#pragma unroll
  for (int e = 0; e < 16; ++e) {
    int k = kb + ((e < 8) ? e : (8 + e));
    float v = (n >= 0 && k < 300) ? src[(size_t)k * N3 + n] : 0.f;
    o[e] = (__bf16)v;
  }
}

// ---------------------------------------------------------------------------
// Pack recurrent weights for scan: dst[g][kt][q][lane][16], q = gate (z,r,h).
// The 3 gate tiles of a (g,kt) are contiguous -> one clause, L2-friendly.
// ---------------------------------------------------------------------------
__global__ void pack_scan(const float* __restrict__ src, __bf16* __restrict__ dst) {
  int idx = blockIdx.x * 256 + threadIdx.x;   // one thread per (g,kt,q,lane)
  if (idx >= NG * KT * 3 * 32) return;
  int lane = idx & 31;
  int q    = (idx >> 5) % 3;
  int kt   = (idx / (32 * 3)) % KT;
  int g    = idx / (32 * 3 * KT);
  int colc = g * 16 + (lane & 15);            // unit index within gate block
  int n    = q * 300 + colc;
  int kb   = kt * 32 + ((lane < 16) ? 0 : 8);
  __bf16* o = dst + (size_t)idx * 16;
#pragma unroll
  for (int e = 0; e < 16; ++e) {
    int k = kb + ((e < 8) ? e : (8 + e));
    float v = (colc < 300 && k < 300) ? src[(size_t)k * N3 + n] : 0.f;
    o[e] = (__bf16)v;
  }
}

// ---------------------------------------------------------------------------
// Input projection: xp[dir][m][n'] = x @ K_dir + b_in_dir, m = b*T+t.
// One wave per (mtile, dir). A (16x320 bf16) stays in registers; ntiles are
// processed 3 at a time with a kt double-buffer, so every load clause feeds
// three independent accumulator chains.
// ---------------------------------------------------------------------------
__global__ void __launch_bounds__(256)
proj(const float* __restrict__ x,
     const __bf16* __restrict__ Bk0, const __bf16* __restrict__ Bk1,
     const float* __restrict__ bin0, const float* __restrict__ bin1,
     float* __restrict__ xp) {
  const int lane = threadIdx.x & 31;
  const int wid  = blockIdx.x * (blockDim.x >> 5) + (threadIdx.x >> 5);
  const int MT   = (B_ * T_) / 16;   // 4096
  const int mtile = wid % MT;
  const int dir   = wid / MT;
  const __bf16* Bp  = dir ? Bk1 : Bk0;
  const float*  bin = dir ? bin1 : bin0;
  float* xpd = xp + (size_t)dir * (size_t)(B_ * T_) * NP;

  const int row = mtile * 16 + (lane & 15);
  const int kb  = (lane < 16) ? 0 : 8;
  const float* xr = x + (size_t)row * D_;
  v16bf a[KT];
#pragma unroll
  for (int kt = 0; kt < KT; ++kt) {
#pragma unroll
    for (int e = 0; e < 16; ++e) {
      int k = kt * 32 + ((e < 8) ? (kb + e) : (16 + kb + (e - 8)));
      float v = (k < D_) ? xr[k] : 0.f;
      a[kt][e] = (__bf16)v;
    }
  }

  const int rb = mtile * 16 + ((lane < 16) ? 0 : 8);     // C layout rows
  for (int nt0 = 0; nt0 < NT; nt0 += 3) {
    const __bf16* bpl = Bp + (size_t)nt0 * (KT * 32 * 16) + (size_t)lane * 16;
    v8f c[3] = {};
    v16bf bb[2][3];
#pragma unroll
    for (int j = 0; j < 3; ++j)
      bb[0][j] = *(const v16bf*)(bpl + (size_t)j * (KT * 32 * 16));
#pragma unroll
    for (int kt = 0; kt < KT; ++kt) {
      if (kt + 1 < KT) {
#pragma unroll
        for (int j = 0; j < 3; ++j)
          bb[(kt + 1) & 1][j] = *(const v16bf*)(bpl + (size_t)j * (KT * 32 * 16)
                                                    + (size_t)(kt + 1) * (32 * 16));
      }
#pragma unroll
      for (int j = 0; j < 3; ++j)
        c[j] = __builtin_amdgcn_wmma_f32_16x16x32_bf16(false, a[kt], false, bb[kt & 1][j],
                                                       (short)0, c[j], false, false);
    }
#pragma unroll
    for (int j = 0; j < 3; ++j) {
      const int npc = (nt0 + j) * 16 + (lane & 15);
      const int n   = np_to_n(npc);
      const float bv = (n >= 0) ? bin[n] : 0.f;
#pragma unroll
      for (int i = 0; i < 8; ++i)
        __builtin_nontemporal_store(c[j][i] + bv, &xpd[(size_t)(rb + i) * NP + npc]);
    }
  }
}

// ---------------------------------------------------------------------------
// Recurrent scan. grid = 8 blocks: (batch-tile 0..3) x (dir 0..1), 512 threads.
// Wave w owns unit-group j = w (plus j = w+16 for w<3). Double-buffered A/B
// feeds 3 interleaved WMMA chains (z,r,h). Gates run entirely in registers via
// the C-fragment (row,unit) mapping. Only 8 blocks run -> occupancy is
// irrelevant; launch_bounds lets the backend keep all buffers in registers.
// ---------------------------------------------------------------------------
__global__ void __launch_bounds__(512)
scan(const float* __restrict__ xp,
     const __bf16* __restrict__ Bs0, const __bf16* __restrict__ Bs1,
     const float* __restrict__ brec0, const float* __restrict__ brec1,
     const unsigned char* __restrict__ mask,
     float* __restrict__ out,    // [B,T,2U]
     float* __restrict__ hfin) { // [2][B][U]
  __shared__ __bf16 h_bf[16][KP];               // 10 KB bf16 shadow of h (A operand)
  __shared__ unsigned char msk[16];

  const int bt   = blockIdx.x & 3;
  const int dir  = blockIdx.x >> 2;
  const int b0   = bt * 16;
  const int lane = threadIdx.x & 31;
  const int w    = threadIdx.x >> 5;            // 0..15
  const __bf16* Bs   = dir ? Bs1 : Bs0;
  const float*  brec = dir ? brec1 : brec0;
  const float*  xpd  = xp + (size_t)dir * (size_t)(B_ * T_) * NP;

  for (int i = threadIdx.x; i < 16 * KP; i += blockDim.x)
    ((__bf16*)h_bf)[i] = (__bf16)0.f;

  const int ngroups = (w < 3) ? 2 : 1;
  int gj[2]; gj[0] = w; gj[1] = w + 16;
  float hf[2][8];
#pragma unroll
  for (int g = 0; g < 2; ++g)
#pragma unroll
    for (int i = 0; i < 8; ++i) hf[g][i] = 0.f;

  const int rlo   = (lane < 16) ? 0 : 8;        // C-layout row base
  const int col   = lane & 15;
  const int arow  = lane & 15;                  // A-layout row
  const int abase = (lane < 16) ? 0 : 8;        // A-layout k-half

  // Stage mask for s = 0.
  if (threadIdx.x < 16) {
    const int t0 = dir ? (T_ - 1) : 0;
    msk[threadIdx.x] = mask[(size_t)(b0 + threadIdx.x) * T_ + t0];
  }
  __syncthreads();

  for (int s = 0; s < T_; ++s) {
    const int t = dir ? (T_ - 1 - s) : s;

    for (int gi = 0; gi < ngroups; ++gi) {
      const int g = gj[gi];
      const int u = g * 16 + col;
      const bool uvalid = (u < U_);
      const __bf16* bg = Bs + (size_t)g * (KT * 3 * 32 * 16) + (size_t)lane * 16;
      v8f cz = {}, cr = {}, ch = {};

      // --- double-buffered A (LDS) + B (global, L2-hot) feeding 3 WMMA chains
      union { v16bf v; v8bf h[2]; } acur, anxt;
      v16bf bcur[3], bnxt[3];
      acur.h[0] = *(const v8bf*)&h_bf[arow][abase];
      acur.h[1] = *(const v8bf*)&h_bf[arow][16 + abase];
      bcur[0] = *(const v16bf*)(bg + 0 * (32 * 16));
      bcur[1] = *(const v16bf*)(bg + 1 * (32 * 16));
      bcur[2] = *(const v16bf*)(bg + 2 * (32 * 16));
#pragma unroll
      for (int kt = 0; kt < KT; ++kt) {
        if (kt + 1 < KT) {
          const int kn = kt + 1;
          anxt.h[0] = *(const v8bf*)&h_bf[arow][kn * 32 + abase];
          anxt.h[1] = *(const v8bf*)&h_bf[arow][kn * 32 + 16 + abase];
          const __bf16* bp = bg + (size_t)kn * (3 * 32 * 16);
          bnxt[0] = *(const v16bf*)(bp + 0 * (32 * 16));
          bnxt[1] = *(const v16bf*)(bp + 1 * (32 * 16));
          bnxt[2] = *(const v16bf*)(bp + 2 * (32 * 16));
        }
        cz = __builtin_amdgcn_wmma_f32_16x16x32_bf16(false, acur.v, false, bcur[0], (short)0, cz, false, false);
        cr = __builtin_amdgcn_wmma_f32_16x16x32_bf16(false, acur.v, false, bcur[1], (short)0, cr, false, false);
        ch = __builtin_amdgcn_wmma_f32_16x16x32_bf16(false, acur.v, false, bcur[2], (short)0, ch, false, false);
        acur.v = anxt.v;
        bcur[0] = bnxt[0]; bcur[1] = bnxt[1]; bcur[2] = bnxt[2];
      }

      float bz_ = 0.f, br_ = 0.f, bh_ = 0.f;
      if (uvalid) { bz_ = brec[u]; br_ = brec[U_ + u]; bh_ = brec[2 * U_ + u]; }
#pragma unroll
      for (int i = 0; i < 8; ++i) {
        const int row  = rlo + i;
        const int bidx = b0 + row;
        float xz = 0.f, xr = 0.f, xh = 0.f;
        if (uvalid) {
          const size_t xrow = ((size_t)bidx * T_ + t) * NP;
          xz = xpd[xrow + u];
          xr = xpd[xrow + 304 + u];
          xh = xpd[xrow + 608 + u];
        }
        const float z  = 1.f / (1.f + __expf(-(xz + cz[i] + bz_)));
        const float rr = 1.f / (1.f + __expf(-(xr + cr[i] + br_)));
        const float hh = tanhf(xh + rr * (ch[i] + bh_));
        const float hn = z * hf[gi][i] + (1.f - z) * hh;
        const bool mm  = msk[row] != 0;
        const float hc = mm ? hn : hf[gi][i];
        hf[gi][i] = hc;
        if (uvalid)
          out[((size_t)bidx * T_ + t) * (2 * U_) + (size_t)dir * U_ + u] = mm ? hn : 0.f;
      }
    }
    __syncthreads();           // all A reads + msk reads of step s complete

    // h shadow update + stage next mask + prefetch xp two steps ahead.
    for (int gi = 0; gi < ngroups; ++gi) {
      const int g = gj[gi];
      const int u = g * 16 + col;
      if (u < U_) {
#pragma unroll
        for (int i = 0; i < 8; ++i)
          h_bf[rlo + i][u] = (__bf16)hf[gi][i];
      }
    }
    if (s + 1 < T_) {
      if (threadIdx.x < 16) {
        const int tn = dir ? (T_ - 2 - s) : (s + 1);
        msk[threadIdx.x] = mask[(size_t)(b0 + threadIdx.x) * T_ + tn];
      }
      if (s + 2 < T_) {
        const int tp   = dir ? (T_ - 3 - s) : (s + 2);
        const int prow = threadIdx.x & 15;
        const int pch  = threadIdx.x >> 4;      // 0..31; 29 x 128B cover a 3648B row
        if (pch < 29)
          __builtin_prefetch(xpd + ((size_t)(b0 + prow) * T_ + tp) * NP + (size_t)pch * 32, 0, 3);
      }
    }
    __syncthreads();
  }

  for (int gi = 0; gi < ngroups; ++gi) {
    const int g = gj[gi];
    const int u = g * 16 + col;
    if (u < U_) {
#pragma unroll
      for (int i = 0; i < 8; ++i)
        hfin[((size_t)dir * B_ + (b0 + rlo + i)) * U_ + u] = hf[gi][i];
    }
  }
}

// ---------------------------------------------------------------------------
// state = relu([h_f | h_b] @ w_fc + b_fc); tiny (64x600x300).
// ---------------------------------------------------------------------------
__global__ void fc(const float* __restrict__ hfin, const float* __restrict__ wfc,
                   const float* __restrict__ bfc, float* __restrict__ st) {
  int idx = blockIdx.x * 256 + threadIdx.x;
  if (idx >= B_ * U_) return;
  int b = idx / U_, o = idx % U_;
  float acc = bfc[o];
  for (int i = 0; i < U_; ++i)
    acc += hfin[(size_t)b * U_ + i] * wfc[(size_t)i * U_ + o];
  for (int i = 0; i < U_; ++i)
    acc += hfin[((size_t)B_ + b) * U_ + i] * wfc[(size_t)(U_ + i) * U_ + o];
  st[idx] = fmaxf(acc, 0.f);
}

extern "C" void kernel_launch(void* const* d_in, const int* in_sizes, int n_in,
                              void* d_out, int out_size, void* d_ws, size_t ws_size,
                              hipStream_t stream) {
  const float* x    = (const float*)d_in[0];
  const unsigned char* mask = (const unsigned char*)d_in[1];
  const float* kf   = (const float*)d_in[2];
  const float* rkf  = (const float*)d_in[3];
  const float* bfwd = (const float*)d_in[4];   // [2,900]
  const float* kb   = (const float*)d_in[5];
  const float* rkb  = (const float*)d_in[6];
  const float* bbwd = (const float*)d_in[7];   // [2,900]
  const float* wfc  = (const float*)d_in[8];
  const float* bfc  = (const float*)d_in[9];
  float* out = (float*)d_out;

  // Workspace carve-up (~481 MB; xp dominates at 2*65536*912*4 B).
  char* ws = (char*)d_ws;
  size_t off = 0;
  auto carve = [&](size_t bytes) {
    void* p = ws + off;
    off = (off + bytes + 255) & ~(size_t)255;
    return p;
  };
  const size_t packBytes  = (size_t)NT * KT * 32 * 16 * sizeof(__bf16);     // 583,680
  const size_t packSBytes = (size_t)NG * KT * 3 * 32 * 16 * sizeof(__bf16); // 583,680
  float*  xp   = (float*) carve((size_t)2 * B_ * T_ * NP * sizeof(float));
  __bf16* Bk0  = (__bf16*)carve(packBytes);
  __bf16* Bk1  = (__bf16*)carve(packBytes);
  __bf16* Bs0  = (__bf16*)carve(packSBytes);
  __bf16* Bs1  = (__bf16*)carve(packSBytes);
  float*  hfin = (float*) carve((size_t)2 * B_ * U_ * sizeof(float));

  const int pElems  = NT * KT * 32;
  const int psElems = NG * KT * 3 * 32;
  pack_b<<<(pElems + 255) / 256, 256, 0, stream>>>(kf, Bk0);
  pack_b<<<(pElems + 255) / 256, 256, 0, stream>>>(kb, Bk1);
  pack_scan<<<(psElems + 255) / 256, 256, 0, stream>>>(rkf, Bs0);
  pack_scan<<<(psElems + 255) / 256, 256, 0, stream>>>(rkb, Bs1);

  // 4096 mtiles * 2 dirs = 8192 waves; 8 waves (256 thr) per block -> 1024 blocks.
  proj<<<1024, 256, 0, stream>>>(x, Bk0, Bk1, bfwd, bbwd, xp);

  // 8 persistent workgroups: (batch tile, dir), 512 threads (16 waves) each.
  scan<<<8, 512, 0, stream>>>(xp, Bs0, Bs1, bfwd + N3, bbwd + N3, mask, out, hfin);

  fc<<<(B_ * U_ + 255) / 256, 256, 0, stream>>>(hfin, wfc, bfc,
                                                out + (size_t)B_ * T_ * 2 * U_);
}
